// DiffLlamaAttention_64544768524528
// MI455X (gfx1250) — matrix operand order repaired and checked by reading, not compiled, not run on hardware
//
#include <hip/hip_runtime.h>
#include <hip/hip_bf16.h>
#include <math.h>

// ---------------------------------------------------------------------------
// DiffLlama attention layer for MI455X (gfx1250, wave32, WMMA + TDM + async LDS)
// B=1, S=2048, HID=2048, H=16, KVH=8, D=128
// ---------------------------------------------------------------------------

#define S_     2048
#define HID_   2048
#define H_     16
#define KVH_   8
#define D_     128
#define NQKV_  4096          // H*D + 2*KVH*D
#define LAMBDA_INIT_ 0.35550906759096927f
#define EPS_   1e-6f
#define INV_SQRT_D_ 0.08838834764831845f

typedef __attribute__((ext_vector_type(16))) __bf16       bf16x16;
typedef __attribute__((ext_vector_type(8)))  float        f32x8;
typedef __attribute__((ext_vector_type(4)))  unsigned int u32x4;
typedef __attribute__((ext_vector_type(8)))  unsigned int u32x8;

union BfPack { u32x4 q[2]; bf16x16 v; };

__device__ __forceinline__ unsigned short f2bf(float x) {
    union { float f; unsigned int u; } c; c.f = x;
    unsigned int r = c.u + 0x7FFFu + ((c.u >> 16) & 1u);   // round-to-nearest-even
    return (unsigned short)(r >> 16);
}

__device__ __forceinline__ bf16x16 pack_bf16(u32x4 a, u32x4 b) {
    BfPack p; p.q[0] = a; p.q[1] = b; return p.v;
}

__device__ __forceinline__ f32x8 wmma_bf16(bf16x16 a, bf16x16 b, f32x8 c) {
    // D = A(16x32 bf16) * B(32x16 bf16) + C(16x16 f32)
    return __builtin_amdgcn_wmma_f32_16x16x32_bf16(false, a, false, b,
                                                   (short)0, c, false, false);
}

__device__ __forceinline__ f32x8 zero8() {
    f32x8 z = {0.f,0.f,0.f,0.f,0.f,0.f,0.f,0.f}; return z;
}

__device__ __forceinline__ unsigned int lds_addr(const void* p) {
    return (unsigned int)(unsigned long long)p;   // low 32 bits = LDS offset
}

// Async global -> LDS copy of 16 bytes (tracked by ASYNCcnt, no VGPR data).
__device__ __forceinline__ void async_ld16(const unsigned short* g, unsigned int lds) {
    asm volatile("global_load_async_to_lds_b128 %0, %1, off"
                 :: "v"(lds), "v"(g) : "memory");
}

// Tensor Data Mover: 2D tile (tile_rows x tile_cols elements of 2 bytes) from a
// row-major tensor with row stride 'stride_elems' into a dense LDS tile.
// Tracked by TENSORcnt.  D# built per CDNA5 ISA 8.3/8.4.
__device__ __forceinline__ void tdm_load_2d_bf16(const unsigned short* gtile,
                                                 unsigned int lds,
                                                 unsigned int tile_cols,
                                                 unsigned int tile_rows,
                                                 unsigned int stride_elems) {
    unsigned long long ga = (unsigned long long)gtile;
    u32x4 g0;
    g0[0] = 1u;                                   // count=1, user-mode flags 0
    g0[1] = lds;                                  // lds_addr (bytes)
    g0[2] = (unsigned int)(ga & 0xFFFFFFFFull);   // global_addr[31:0]
    g0[3] = (unsigned int)((ga >> 32) & 0x01FFFFFFull) | 0x80000000u; // addr[56:32] | type=2
    u32x8 g1;
    g1[0] = 0x00010000u;                          // workgroup_mask=0, data_size=1 (2 bytes)
    g1[1] = (stride_elems & 0xFFFFu) << 16;       // tensor_dim0[15:0]
    g1[2] = (stride_elems >> 16) | ((tile_rows & 0xFFFFu) << 16); // dim0 hi | dim1 lo
    g1[3] = (tile_rows >> 16) | (tile_cols << 16);                // dim1 hi | tile_dim0
    g1[4] = tile_rows & 0xFFFFu;                  // tile_dim1 | tile_dim2=0
    g1[5] = stride_elems;                         // tensor_dim0_stride[31:0] (elements)
    g1[6] = 0u;                                   // stride hi | dim1_stride lo
    g1[7] = 0u;                                   // dim1_stride hi
    asm volatile("tensor_load_to_lds %0, %1" :: "s"(g0), "s"(g1) : "memory");
}

// ---------------------------------------------------------------------------
// Converters / repackers
// ---------------------------------------------------------------------------
__global__ void k_cvt_f2bf(const float* __restrict__ src,
                           unsigned short* __restrict__ dst, int n) {
    int i = blockIdx.x * blockDim.x + threadIdx.x;
    if (i < n) dst[i] = f2bf(src[i]);
}

// src [rows][cols] fp32  ->  dst [cols][rows] bf16   (weights to [N][K])
__global__ void k_transpose_f2bf(const float* __restrict__ src,
                                 unsigned short* __restrict__ dst,
                                 int rows, int cols) {
    int i = blockIdx.x * blockDim.x + threadIdx.x;
    if (i < rows * cols) {
        int r = i / cols, c = i - r * cols;
        dst[(size_t)c * rows + r] = f2bf(src[i]);
    }
}

// RoPE on Q/K (Llama half-rotation), repack to WMMA-friendly layouts.
// qkv: [S][4096] fp32 (Q | K | V).  Qbf [H][S][D], Kbf [KVH][S][D], Vt [KVH][D][S]
__global__ void k_rope_pack(const float* __restrict__ qkv,
                            const float* __restrict__ cosb,  // [S][D]
                            const float* __restrict__ sinb,  // [S][D]
                            unsigned short* __restrict__ Qbf,
                            unsigned short* __restrict__ Kbf,
                            unsigned short* __restrict__ Vt) {
    int i = blockIdx.x * blockDim.x + threadIdx.x;
    if (i >= S_ * NQKV_) return;
    int s = i >> 12, col = i & (NQKV_ - 1);
    float x = qkv[i];
    if (col < H_ * D_) {
        int h = col >> 7, d = col & 127;
        float other = (d < 64) ? -qkv[i + 64] : qkv[i - 64];
        float r = x * cosb[s * D_ + d] + other * sinb[s * D_ + d];
        Qbf[((size_t)h * S_ + s) * D_ + d] = f2bf(r);
    } else if (col < H_ * D_ + KVH_ * D_) {
        int c = col - H_ * D_;
        int kv = c >> 7, d = c & 127;
        float other = (d < 64) ? -qkv[i + 64] : qkv[i - 64];
        float r = x * cosb[s * D_ + d] + other * sinb[s * D_ + d];
        Kbf[((size_t)kv * S_ + s) * D_ + d] = f2bf(r);
    } else {
        int c = col - (H_ * D_ + KVH_ * D_);
        int kv = c >> 7, d = c & 127;
        Vt[((size_t)kv * D_ + d) * S_ + s] = f2bf(x);
    }
}

// ---------------------------------------------------------------------------
// bf16 GEMM, double-buffered LDS staging:
//   A-tile (128x32) fetched by the Tensor Data Mover (TENSORcnt),
//   B-tile (128x32) fetched by async global->LDS b128 copies (ASYNCcnt).
//   C[M][N] (f32) = A[M][K] (bf16, row-major) * Bt[N][K]^T
// grid = (M/128, N/128), block = 256 (8 waves); each wave: 16 rows x 128 cols.
// ---------------------------------------------------------------------------
__global__ __launch_bounds__(256)
void k_gemm_bf16(const unsigned short* __restrict__ A,
                 const unsigned short* __restrict__ Bt,
                 float* __restrict__ C, int M, int N, int K) {
    __shared__ __align__(16) unsigned short As[2][128][32];
    __shared__ __align__(16) unsigned short Bs[2][128][32];

    const int tid  = threadIdx.x;
    const int wave = tid >> 5;
    const int lane = tid & 31;
    const int hi   = lane >> 4;          // lane half (0/1)
    const int ln   = lane & 15;
    const int rowg = blockIdx.x * 128;
    const int colg = blockIdx.y * 128;

    f32x8 acc[8];
#pragma unroll
    for (int t = 0; t < 8; ++t) acc[t] = zero8();

    // A tile via TDM (wave 0 only), B tile cooperatively via async loads.
    auto issue = [&](int kc, int buf) {
        const int k0 = kc * 32;
        if (wave == 0)
            tdm_load_2d_bf16(A + (size_t)rowg * K + k0,
                             lds_addr(&As[buf][0][0]), 32u, 128u, (unsigned)K);
#pragma unroll
        for (int s = 0; s < 2; ++s) {
            int seg  = tid + s * 256;            // 512 x 16B segments
            int row  = seg >> 2;
            int part = (seg & 3) * 8;
            async_ld16(Bt + (size_t)(colg + row) * K + k0 + part,
                       lds_addr(&Bs[buf][row][part]));
        }
    };

    const int nk = K >> 5;
    issue(0, 0);
    for (int kc = 0; kc < nk; ++kc) {
        const int cur = kc & 1;
        if (kc + 1 < nk) {
            issue(kc + 1, cur ^ 1);
            asm volatile("s_wait_asynccnt 0x2" ::: "memory");   // chunk kc B landed
            if (wave == 0) __builtin_amdgcn_s_wait_tensorcnt(1); // chunk kc A landed
        } else {
            asm volatile("s_wait_asynccnt 0x0" ::: "memory");
            if (wave == 0) __builtin_amdgcn_s_wait_tensorcnt(0);
        }
        __syncthreads();   // all waves' portions of chunk kc visible

        const unsigned short* ar = &As[cur][wave * 16 + ln][0];
        u32x4 a0 = *(const u32x4*)(const void*)(ar + hi * 16);
        u32x4 a1 = *(const u32x4*)(const void*)(ar + hi * 16 + 8);
        bf16x16 av = pack_bf16(a0, a1);

        // batch all B-operand LDS loads, then stream the 8 WMMAs
        u32x4 bl[8], bh[8];
#pragma unroll
        for (int t = 0; t < 8; ++t) {
            const unsigned short* br = &Bs[cur][t * 16 + ln][0];
            bl[t] = *(const u32x4*)(const void*)(br + hi * 16);
            bh[t] = *(const u32x4*)(const void*)(br + hi * 16 + 8);
        }
#pragma unroll
        for (int t = 0; t < 8; ++t)
            acc[t] = wmma_bf16(av, pack_bf16(bl[t], bh[t]), acc[t]);

        __syncthreads();   // reads of 'cur' done before it is refilled
    }

#pragma unroll
    for (int t = 0; t < 8; ++t) {
        int col = colg + t * 16 + ln;
#pragma unroll
        for (int r = 0; r < 8; ++r) {
            int row = rowg + wave * 16 + r + 8 * hi;   // C layout: M = r + 8*hi
            C[(size_t)row * N + col] = acc[t][r];
        }
    }
}

// ---------------------------------------------------------------------------
// Attention: one wave per (head, 16-query strip).
// Pass 1: WMMA scores -> row max.  Pass 2: WMMA scores again, p = exp(s-max),
// write unnormalized p, stage p through LDS, O += P * V_cat via WMMA.
// ---------------------------------------------------------------------------
__global__ __launch_bounds__(32)
void k_attn(const unsigned short* __restrict__ Qbf,   // [H][S][D]
            const unsigned short* __restrict__ Kbf,   // [KVH][S][D]
            const unsigned short* __restrict__ Vt,    // [KVH][D][S]
            float* __restrict__ Pout,                 // [H][S][S] unnormalized
            float* __restrict__ Ounn,                 // [H][S][256]
            float* __restrict__ Rsum) {               // [H][S]
    __shared__ __align__(16) unsigned short pls[16 * 32];  // P tile staging

    const int h    = blockIdx.x;
    const int qb   = blockIdx.y * 16;
    const int lane = threadIdx.x & 31;
    const int hi   = lane >> 4;
    const int ln   = lane & 15;
    const int kv   = h >> 1;                  // GQA: K head
    const int kend = qb + 16;                 // causal: keys [0, kend)

    // Q A-operands: 4 chunks over d (K-dim of WMMA)
    const unsigned short* qrow = Qbf + ((size_t)h * S_ + (qb + ln)) * D_;
    bf16x16 qa[4];
#pragma unroll
    for (int c = 0; c < 4; ++c) {
        u32x4 a0 = *(const u32x4*)(const void*)(qrow + c * 32 + hi * 16);
        u32x4 a1 = *(const u32x4*)(const void*)(qrow + c * 32 + hi * 16 + 8);
        qa[c] = pack_bf16(a0, a1);
    }

    // one 16x16 score tile vs keys [kbase, kbase+16): batch loads, then WMMA chain
    auto score_tile = [&](int kbase) -> f32x8 {
        const unsigned short* krow = Kbf + ((size_t)kv * S_ + (kbase + ln)) * D_;
        u32x4 b0[4], b1[4];
#pragma unroll
        for (int c = 0; c < 4; ++c) {
            b0[c] = *(const u32x4*)(const void*)(krow + c * 32 + hi * 16);
            b1[c] = *(const u32x4*)(const void*)(krow + c * 32 + hi * 16 + 8);
        }
        f32x8 sc = zero8();
#pragma unroll
        for (int c = 0; c < 4; ++c)
            sc = wmma_bf16(qa[c], pack_bf16(b0[c], b1[c]), sc);
        return sc;
    };

    // ---- pass 1: row max of scaled causal scores ----
    float rmax[8];
#pragma unroll
    for (int r = 0; r < 8; ++r) rmax[r] = -1e30f;

    for (int kb = 0; kb < kend; kb += 16) {
        __builtin_prefetch(Kbf + ((size_t)kv * S_ + (kb + 16 + ln)) * D_, 0, 1);
        f32x8 sc = score_tile(kb);
#pragma unroll
        for (int r = 0; r < 8; ++r) {
            int qg = qb + r + 8 * hi, kc = kb + ln;
            float v = (kc <= qg) ? sc[r] * INV_SQRT_D_ : -1e30f;
            rmax[r] = fmaxf(rmax[r], v);
        }
    }
#pragma unroll
    for (int m = 1; m < 16; m <<= 1)
#pragma unroll
        for (int r = 0; r < 8; ++r)
            rmax[r] = fmaxf(rmax[r], __shfl_xor(rmax[r], m, 16));

    // ---- pass 2: exp, store P, accumulate O = P * V_cat ----
    f32x8 o[16];
#pragma unroll
    for (int t = 0; t < 16; ++t) o[t] = zero8();
    float psum[8];
#pragma unroll
    for (int r = 0; r < 8; ++r) psum[r] = 0.f;

    for (int kb = 0; kb < kend; kb += 32) {
        float pt[2][8];
#pragma unroll
        for (int s2 = 0; s2 < 2; ++s2) {
            int kbase = kb + s2 * 16;
            f32x8 sc = zero8();
            if (kbase < kend) sc = score_tile(kbase);
#pragma unroll
            for (int r = 0; r < 8; ++r) {
                int qg = qb + r + 8 * hi, kc = kbase + ln;
                float p = 0.f;
                if (kbase < kend && kc <= qg)
                    p = __expf(sc[r] * INV_SQRT_D_ - rmax[r]);
                pt[s2][r] = p;
                psum[r] += p;
                if (kbase < kend)
                    Pout[((size_t)h * S_ + qg) * S_ + kc] = p;
            }
        }

        // stage P (16 rows x 32 keys) into LDS as bf16, row-major
        __syncthreads();
#pragma unroll
        for (int s2 = 0; s2 < 2; ++s2)
#pragma unroll
            for (int r = 0; r < 8; ++r)
                pls[(r + 8 * hi) * 32 + s2 * 16 + ln] = f2bf(pt[s2][r]);
        __syncthreads();

        // reload as A operand (16x32)
        u32x4 p0 = *(const u32x4*)(const void*)(pls + ln * 32 + hi * 16);
        u32x4 p1 = *(const u32x4*)(const void*)(pls + ln * 32 + hi * 16 + 8);
        bf16x16 pa = pack_bf16(p0, p1);

        // O (16 x 256) += P * V_cat; batch V loads in groups of 4 tiles
#pragma unroll
        for (int g = 0; g < 4; ++g) {
            u32x4 vb0[4], vb1[4];
#pragma unroll
            for (int u = 0; u < 4; ++u) {
                int t  = g * 4 + u;
                int f0 = t * 16;
                int vsel = (f0 < 128) ? kv : (4 + kv);
                int d    = (f0 < 128) ? (f0 + ln) : (f0 - 128 + ln);
                const unsigned short* vrow = Vt + ((size_t)vsel * D_ + d) * S_;
                vb0[u] = *(const u32x4*)(const void*)(vrow + kb + hi * 16);
                vb1[u] = *(const u32x4*)(const void*)(vrow + kb + hi * 16 + 8);
            }
#pragma unroll
            for (int u = 0; u < 4; ++u) {
                int t = g * 4 + u;
                o[t] = wmma_bf16(pa, pack_bf16(vb0[u], vb1[u]), o[t]);
            }
        }
    }

    // row sums across the 16 lanes of each half
#pragma unroll
    for (int m = 1; m < 16; m <<= 1)
#pragma unroll
        for (int r = 0; r < 8; ++r) psum[r] += __shfl_xor(psum[r], m, 16);
    if (ln == 0)
#pragma unroll
        for (int r = 0; r < 8; ++r)
            Rsum[(size_t)h * S_ + qb + r + 8 * hi] = psum[r];

#pragma unroll
    for (int t = 0; t < 16; ++t)
#pragma unroll
        for (int r = 0; r < 8; ++r)
            Ounn[((size_t)h * S_ + (qb + r + 8 * hi)) * 256 + t * 16 + ln] = o[t][r];
}

// ---------------------------------------------------------------------------
// Differential combine + RMSNorm -> bf16 activations [S][2048] for final GEMM
// ---------------------------------------------------------------------------
__global__ __launch_bounds__(256)
void k_combine(const float* __restrict__ Ounn, const float* __restrict__ Rsum,
               const float* __restrict__ lq1, const float* __restrict__ lk1,
               const float* __restrict__ lq2, const float* __restrict__ lk2,
               unsigned short* __restrict__ Obf) {
    int idx = blockIdx.x * blockDim.x + threadIdx.x;   // over 8 * S
    if (idx >= (H_ / 2) * S_) return;
    int p = idx >> 11, s = idx & (S_ - 1);

    float d1 = 0.f, d2 = 0.f;
    for (int i = 0; i < D_; ++i) { d1 += lq1[i] * lk1[i]; d2 += lq2[i] * lk2[i]; }
    float lam = __expf(d1) - __expf(d2) + LAMBDA_INIT_;

    const float* o1 = Ounn + ((size_t)p * S_ + s) * 256;
    const float* o2 = Ounn + ((size_t)(p + 8) * S_ + s) * 256;
    float inv1 = 1.f / Rsum[(size_t)p * S_ + s];
    float inv2 = 1.f / Rsum[(size_t)(p + 8) * S_ + s];

    float ss = 0.f;
    for (int f = 0; f < 256; ++f) {
        float v = o1[f] * inv1 - lam * (o2[f] * inv2);
        ss += v * v;
    }
    float rn = rsqrtf(ss * (1.f / 256.f) + EPS_) * (1.f - LAMBDA_INIT_);
    unsigned short* dst = Obf + (size_t)s * HID_ + p * 256;
    for (int f = 0; f < 256; ++f) {
        float v = o1[f] * inv1 - lam * (o2[f] * inv2);
        dst[f] = f2bf(v * rn);
    }
}

// ---------------------------------------------------------------------------
// Normalize attn weights row-by-row; zero the causal upper triangle
// ---------------------------------------------------------------------------
__global__ __launch_bounds__(256)
void k_norm_attn(float* __restrict__ P, const float* __restrict__ Rsum) {
    int rowid = blockIdx.x;               // h * S + q
    int q = rowid & (S_ - 1);
    float inv = 1.f / Rsum[rowid];
    float* row = P + (size_t)rowid * S_;
    for (int c = threadIdx.x; c < S_; c += 256)
        row[c] = (c <= q) ? row[c] * inv : 0.f;
}

// ---------------------------------------------------------------------------
// Host-side launch
// ---------------------------------------------------------------------------
extern "C" void kernel_launch(void* const* d_in, const int* in_sizes, int n_in,
                              void* d_out, int out_size, void* d_ws, size_t ws_size,
                              hipStream_t stream) {
    const float* X    = (const float*)d_in[0];
    const float* cosb = (const float*)d_in[1];
    const float* sinb = (const float*)d_in[2];
    const float* Wq   = (const float*)d_in[3];
    const float* Wk   = (const float*)d_in[4];
    const float* Wv   = (const float*)d_in[5];
    const float* Wo   = (const float*)d_in[6];
    const float* lq1  = (const float*)d_in[7];
    const float* lk1  = (const float*)d_in[8];
    const float* lq2  = (const float*)d_in[9];
    const float* lk2  = (const float*)d_in[10];

    float* out  = (float*)d_out;                         // [S][HID]
    float* Pout = (float*)d_out + (size_t)S_ * HID_;     // [H][S][S]

    char* w = (char*)d_ws;
    const size_t MB = 1024ull * 1024ull;
    unsigned short* Xbf  = (unsigned short*)(w + 0);          //  8 MB
    unsigned short* Wt   = (unsigned short*)(w + 8 * MB);     // 16 MB  [4096][2048]
    unsigned short* Wot  = (unsigned short*)(w + 24 * MB);    //  8 MB  [2048][2048]
    float*          QKV  = (float*)(w + 32 * MB);             // 32 MB  [S][4096]
    unsigned short* Qbf  = (unsigned short*)(w + 64 * MB);    //  8 MB
    unsigned short* Kbf  = (unsigned short*)(w + 72 * MB);    //  4 MB
    unsigned short* Vt   = (unsigned short*)(w + 76 * MB);    //  4 MB
    float*          Ounn = (float*)(w + 80 * MB);             // 32 MB  [16][S][256]
    float*          Rsum = (float*)(w + 112 * MB);            // 128 KB
    unsigned short* Obf  = (unsigned short*)(w + 113 * MB);   //  8 MB  [S][2048]

    // 1) converts / transposes
    {
        int n = S_ * HID_;
        k_cvt_f2bf<<<(n + 255) / 256, 256, 0, stream>>>(X, Xbf, n);
        k_transpose_f2bf<<<(HID_ * (H_ * D_) + 255) / 256, 256, 0, stream>>>(
            Wq, Wt, HID_, H_ * D_);
        k_transpose_f2bf<<<(HID_ * (KVH_ * D_) + 255) / 256, 256, 0, stream>>>(
            Wk, Wt + (size_t)(H_ * D_) * HID_, HID_, KVH_ * D_);
        k_transpose_f2bf<<<(HID_ * (KVH_ * D_) + 255) / 256, 256, 0, stream>>>(
            Wv, Wt + (size_t)(H_ * D_ + KVH_ * D_) * HID_, HID_, KVH_ * D_);
        k_transpose_f2bf<<<(HID_ * HID_ + 255) / 256, 256, 0, stream>>>(
            Wo, Wot, H_ * D_, HID_);
    }

    // 2) QKV projection GEMM: [S][HID] x [HID][4096] -> [S][4096] f32
    k_gemm_bf16<<<dim3(S_ / 128, NQKV_ / 128), 256, 0, stream>>>(
        Xbf, Wt, QKV, S_, NQKV_, HID_);

    // 3) RoPE + repack
    k_rope_pack<<<(S_ * NQKV_ + 255) / 256, 256, 0, stream>>>(
        QKV, cosb, sinb, Qbf, Kbf, Vt);

    // 4) attention (scores + softmax stats + P*V), writes unnormalized P
    k_attn<<<dim3(H_, S_ / 16), 32, 0, stream>>>(Qbf, Kbf, Vt, Pout, Ounn, Rsum);

    // 5) differential combine + RMSNorm -> bf16 activations
    k_combine<<<((H_ / 2) * S_ + 255) / 256, 256, 0, stream>>>(
        Ounn, Rsum, lq1, lk1, lq2, lk2, Obf);

    // 6) finalize attn_weights (normalize rows, zero upper triangle)
    k_norm_attn<<<H_ * S_, 256, 0, stream>>>(Pout, Rsum);

    // 7) output projection: [S][2048] x [2048][2048] -> d_out f32
    k_gemm_bf16<<<dim3(S_ / 128, HID_ / 128), 256, 0, stream>>>(
        Obf, Wot, out, S_, HID_, HID_);
}
